// StARformer_78477642432846
// MI455X (gfx1250) — matrix operand order repaired and checked by reading, not compiled, not run on hardware
//
#include <hip/hip_runtime.h>
#include <cstdint>
#include <cstddef>

// ---------------- model dimensions ----------------
#define Bsz  32
#define Nst  32
#define Mloc 64
#define NLd  192
#define NGd  768
#define HLh  6
#define HGh  12
#define T1   (Bsz * Nst * Mloc)   // 65536 local tokens
#define SEQ1 (Bsz * Nst)          // 1024 local sequences
#define LGl  96                   // 3*Nst global seq length
#define T2   (Bsz * LGl)          // 3072 global tokens

typedef unsigned short u16;
typedef __attribute__((ext_vector_type(16))) __bf16 v16bf;
typedef __attribute__((ext_vector_type(8)))  float  v8f;

// ---------------- bf16 helpers (raw u16 storage) ----------------
static __device__ __forceinline__ u16 f2bf_raw(float f) {
    union { float f; unsigned u; } c; c.f = f;
    unsigned u = c.u;
    return (u16)((u + 0x7FFFu + ((u >> 16) & 1u)) >> 16);
}
static __device__ __forceinline__ float bf2f(u16 r) {
    union { unsigned u; float f; } c; c.u = ((unsigned)r) << 16; return c.f;
}

// ---------------- vectorized WMMA fragment loaders ----------------------
// Per CDNA5 ISA 16-bit layouts. All source addresses are 16B
// aligned by construction, so these lower to ds_load_b128 pairs.
union FragU { v16bf v; uint4 q[2]; };

// A fragment 16x32: source row-major [m][k]; lane needs k runs kb..kb+7 and
// kb+16..kb+23 of row m (kb = 0 or 8 by half-wave) -> two contiguous uint4.
static __device__ __forceinline__ v16bf load_frag_a_vec(const u16* s, int rowStride,
                                                        int mBase, int kBase, int lane) {
    int m  = mBase + (lane & 15);
    int kb = kBase + ((lane < 16) ? 0 : 8);
    const u16* p = s + (size_t)m * rowStride + kb;
    FragU f;
    f.q[0] = *(const uint4*)(p);
    f.q[1] = *(const uint4*)(p + 16);
    return f.v;
}
// B fragment 32x16: source stored [n][k]; lane needs k runs kb..kb+7 and
// kb+16..kb+23 of row n (kb = 0 or 8 by half-wave) -> two contiguous uint4.
static __device__ __forceinline__ v16bf load_frag_bk_vec(const u16* s, int rowStride,
                                                         int nBase, int kBase, int lane) {
    int n  = nBase + (lane & 15);
    int kb = kBase + ((lane < 16) ? 0 : 8);
    const u16* p = s + (size_t)n * rowStride + kb;
    FragU f;
    f.q[0] = *(const uint4*)(p);
    f.q[1] = *(const uint4*)(p + 16);
    return f.v;
}

// ---------------- LayerNorm: fp32 in -> bf16 out ----------------
__global__ void __launch_bounds__(256)
ln_bf16_kernel(const float* __restrict__ x, const float* __restrict__ g,
               const float* __restrict__ b, u16* __restrict__ out, int D) {
    __shared__ float red[256];
    int row = blockIdx.x;
    const float* xr = x + (size_t)row * D;
    float s = 0.f;
    for (int i = threadIdx.x; i < D; i += 256) s += xr[i];
    red[threadIdx.x] = s; __syncthreads();
    for (int st = 128; st > 0; st >>= 1) {
        if ((int)threadIdx.x < st) red[threadIdx.x] += red[threadIdx.x + st];
        __syncthreads();
    }
    float mean = red[0] / (float)D;
    __syncthreads();
    float vs = 0.f;
    for (int i = threadIdx.x; i < D; i += 256) { float d = xr[i] - mean; vs += d * d; }
    red[threadIdx.x] = vs; __syncthreads();
    for (int st = 128; st > 0; st >>= 1) {
        if ((int)threadIdx.x < st) red[threadIdx.x] += red[threadIdx.x + st];
        __syncthreads();
    }
    float inv = rsqrtf(red[0] / (float)D + 1e-5f);
    for (int i = threadIdx.x; i < D; i += 256)
        out[(size_t)row * D + i] = f2bf_raw((xr[i] - mean) * inv * g[i] + b[i]);
}

// ---------------- generic bf16 WMMA GEMM -------------------------------
// C[M,N] = epi( A_bf16[M,K] @ Bw_f32[K,N] + bias ) [+ resid], epi may gelu.
// Block tile 128x64, 8 waves as 4x2; each wave owns a 32x32 sub-tile =
// 2x2 WMMA tiles, so one K-step = 4 fragment loads (8 ds_load_b128) feeding
// 4 v_wmma ops. A tile staged via async global->LDS b128 (2 per lane, 8KB);
// B tile f32->bf16, stored transposed [n][k] for contiguous fragments.
#define GBM 128
#define GBN 64
#define GBK 32
__global__ void __launch_bounds__(256)
gemm_bf16_kernel(const u16* __restrict__ A, const float* __restrict__ Bw,
                 const float* __restrict__ bias, const float* __restrict__ resid,
                 float* __restrict__ outF, u16* __restrict__ outB,
                 int Mdim, int Ndim, int Kdim, int geluFlag) {
    __shared__ u16 As [GBM][GBK];   // [m][k]  8 KB
    __shared__ u16 BsT[GBN][GBK];   // [n][k]  4 KB
    const int tid  = threadIdx.x;
    const int lane = tid & 31;
    const int wave = tid >> 5;
    const int wm = wave & 3, wn = wave >> 2;   // wave sub-tile: rows wm*32, cols wn*32
    const int rowBase = blockIdx.y * GBM;
    const int colBase = blockIdx.x * GBN;
    // per-thread staging coordinates
    const int ar = tid >> 1;             // A: 128 rows, 32B (2 x b128) per thread
    const int ac = (tid & 1) * 16;
    const int br = tid >> 3;             // B: 32 k-rows, 8 floats (2 x float4)/thread
    const int bc = (tid & 7) * 8;
    v8f acc00 = {0.f,0.f,0.f,0.f,0.f,0.f,0.f,0.f};
    v8f acc01 = {0.f,0.f,0.f,0.f,0.f,0.f,0.f,0.f};
    v8f acc10 = {0.f,0.f,0.f,0.f,0.f,0.f,0.f,0.f};
    v8f acc11 = {0.f,0.f,0.f,0.f,0.f,0.f,0.f,0.f};
    for (int k0 = 0; k0 < Kdim; k0 += GBK) {
        // ---- A tile: async copy global -> LDS, 2 x 16B per lane (8KB total)
        {
            const u16* gp = A + (size_t)(rowBase + ar) * Kdim + k0 + ac;
            unsigned lds0 = (unsigned)(uintptr_t)&As[ar][ac];
            unsigned long long ga0 = (unsigned long long)(uintptr_t)gp;
            asm volatile("global_load_async_to_lds_b128 %0, %1, off"
                         :: "v"(lds0), "v"(ga0) : "memory");
            unsigned lds1 = (unsigned)(uintptr_t)&As[ar][ac + 8];
            unsigned long long ga1 = (unsigned long long)(uintptr_t)(gp + 8);
            asm volatile("global_load_async_to_lds_b128 %0, %1, off"
                         :: "v"(lds1), "v"(ga1) : "memory");
        }
        // ---- B tile: two float4 loads, convert, store transposed
        {
            const float* bp = Bw + (size_t)(k0 + br) * Ndim + colBase + bc;
            const float4 b0 = *(const float4*)(bp);
            const float4 b1 = *(const float4*)(bp + 4);
            BsT[bc + 0][br] = f2bf_raw(b0.x);
            BsT[bc + 1][br] = f2bf_raw(b0.y);
            BsT[bc + 2][br] = f2bf_raw(b0.z);
            BsT[bc + 3][br] = f2bf_raw(b0.w);
            BsT[bc + 4][br] = f2bf_raw(b1.x);
            BsT[bc + 5][br] = f2bf_raw(b1.y);
            BsT[bc + 6][br] = f2bf_raw(b1.z);
            BsT[bc + 7][br] = f2bf_raw(b1.w);
        }
        if (k0 + GBK < Kdim)
            __builtin_prefetch(&Bw[(size_t)(k0 + GBK + br) * Ndim + colBase + bc], 0, 1);
        asm volatile("s_wait_asynccnt 0" ::: "memory");
        __syncthreads();
        v16bf a0 = load_frag_a_vec(&As[0][0],  GBK, wm * 32,      0, lane);
        v16bf a1 = load_frag_a_vec(&As[0][0],  GBK, wm * 32 + 16, 0, lane);
        v16bf b0 = load_frag_bk_vec(&BsT[0][0], GBK, wn * 32,      0, lane);
        v16bf b1 = load_frag_bk_vec(&BsT[0][0], GBK, wn * 32 + 16, 0, lane);
        acc00 = __builtin_amdgcn_wmma_f32_16x16x32_bf16(false, a0, false, b0,
                                                        (short)0, acc00, false, false);
        acc01 = __builtin_amdgcn_wmma_f32_16x16x32_bf16(false, a0, false, b1,
                                                        (short)0, acc01, false, false);
        acc10 = __builtin_amdgcn_wmma_f32_16x16x32_bf16(false, a1, false, b0,
                                                        (short)0, acc10, false, false);
        acc11 = __builtin_amdgcn_wmma_f32_16x16x32_bf16(false, a1, false, b1,
                                                        (short)0, acc11, false, false);
        __syncthreads();
    }
    // ---- epilogue: 2x2 tiles per wave
    const v8f* accs[4] = { &acc00, &acc01, &acc10, &acc11 };
#pragma unroll
    for (int ti = 0; ti < 2; ++ti) {
#pragma unroll
        for (int tj = 0; tj < 2; ++tj) {
            const v8f& a = *accs[ti * 2 + tj];
            const int n    = colBase + wn * 32 + tj * 16 + (lane & 15);
            const int mOff = rowBase + wm * 32 + ti * 16 + ((lane < 16) ? 0 : 8);
#pragma unroll
            for (int r = 0; r < 8; ++r) {
                int mm = mOff + r;
                float v = a[r];
                if (bias) v += bias[n];
                if (geluFlag) v = 0.5f * v * (1.0f + erff(v * 0.70710678118654752f));
                size_t o = (size_t)mm * Ndim + n;
                if (resid) v += resid[o];
                if (outF) outF[o] = v;
                if (outB) outB[o] = f2bf_raw(v);
            }
        }
    }
}

// ---------------- fused causal attention per (seq, head) ----------------
// q,k,v,y: bf16 [numSeq*L, nh*hd].
// Dynamic LDS (all u16): Ssc[L*L] scores->P in place, qs[L][hd], ks[L][hd]
// (row-major: the k^T fragment is contiguous in k), vsT[hd][L] (transposed so
// the V fragment is contiguous in k).
__global__ void __launch_bounds__(256)
attn_kernel(const u16* __restrict__ q, const u16* __restrict__ k,
            const u16* __restrict__ v, u16* __restrict__ y,
            int L, int hd, int nh, float scale) {
    extern __shared__ unsigned char smem[];
    u16* Ssc = (u16*)smem;
    u16* qs  = Ssc + (size_t)L * L;
    u16* ks  = qs  + (size_t)L * hd;
    u16* vsT = ks  + (size_t)L * hd;
    const int D = nh * hd;
    const int seq  = blockIdx.x / nh;
    const int head = blockIdx.x % nh;
    const int tid = threadIdx.x, lane = tid & 31, wave = tid >> 5;

    // ---- stage q/k/v: uint4 (16B) global loads; v transposed into LDS
    const int chunks = (L * hd) / 8;
    for (int i = tid; i < chunks; i += 256) {
        int e0 = i * 8;
        int r = e0 / hd, c = e0 % hd;
        size_t gi = ((size_t)(seq * L + r)) * D + head * hd + c;
        uint4 qq = *(const uint4*)(q + gi);
        uint4 kk = *(const uint4*)(k + gi);
        uint4 vv = *(const uint4*)(v + gi);
        *(uint4*)(qs + e0) = qq;
        *(uint4*)(ks + e0) = kk;
        const u16* vp = (const u16*)&vv;
#pragma unroll
        for (int j = 0; j < 8; ++j) vsT[(size_t)(c + j) * L + r] = vp[j];
    }
    __syncthreads();

    const int nT = L >> 4;
    for (int t = wave; t < nT * nT; t += 8) {          // S = scale * q k^T
        int tm = t / nT, tn = t % nT;
        if (tn * 16 > tm * 16 + 15) continue;          // fully masked tile
        v8f acc = {0.f, 0.f, 0.f, 0.f, 0.f, 0.f, 0.f, 0.f};
        for (int kk = 0; kk < hd; kk += 32) {
            v16bf a = load_frag_a_vec(qs, hd, tm * 16, kk, lane);
            v16bf b = load_frag_bk_vec(ks, hd, tn * 16, kk, lane);
            acc = __builtin_amdgcn_wmma_f32_16x16x32_bf16(false, a, false, b,
                                                          (short)0, acc, false, false);
        }
        int col  = tn * 16 + (lane & 15);
        int rOff = tm * 16 + ((lane < 16) ? 0 : 8);
#pragma unroll
        for (int r = 0; r < 8; ++r) {
            int i = rOff + r;
            if (col <= i) Ssc[(size_t)i * L + col] = f2bf_raw(acc[r] * scale);
        }
    }
    __syncthreads();

    for (int i = tid; i < L; i += 256) {               // causal softmax (bf16 rows)
        float mx = -3.0e38f;
        for (int j = 0; j <= i; ++j) mx = fmaxf(mx, bf2f(Ssc[(size_t)i * L + j]));
        float sum = 0.f;
        for (int j = 0; j <= i; ++j) {
            float e = __expf(bf2f(Ssc[(size_t)i * L + j]) - mx);
            Ssc[(size_t)i * L + j] = f2bf_raw(e); sum += e;
        }
        float inv = 1.f / sum;
        for (int j = 0; j < L; ++j)
            Ssc[(size_t)i * L + j] =
                (j <= i) ? f2bf_raw(bf2f(Ssc[(size_t)i * L + j]) * inv) : (u16)0;
    }
    __syncthreads();

    const int nTn = hd >> 4;
    for (int t = wave; t < nT * nTn; t += 8) {         // y = P @ v
        int tm = t / nTn, tn = t % nTn;
        v8f acc = {0.f, 0.f, 0.f, 0.f, 0.f, 0.f, 0.f, 0.f};
        for (int kk = 0; kk < L; kk += 32) {
            v16bf a = load_frag_a_vec(Ssc, L, tm * 16, kk, lane);
            v16bf b = load_frag_bk_vec(vsT, L, tn * 16, kk, lane);
            acc = __builtin_amdgcn_wmma_f32_16x16x32_bf16(false, a, false, b,
                                                          (short)0, acc, false, false);
        }
        int col  = head * hd + tn * 16 + (lane & 15);
        int rOff = tm * 16 + ((lane < 16) ? 0 : 8);
#pragma unroll
        for (int r = 0; r < 8; ++r)
            y[((size_t)(seq * L + rOff + r)) * D + col] = f2bf_raw(acc[r]);
    }
}

// ---------------- interleave zg with xg pairs -> global stream ----------
__global__ void __launch_bounds__(256)
interleave_kernel(const float* __restrict__ zg, const float* __restrict__ xg,
                  float* __restrict__ gx) {
    size_t idx = (size_t)blockIdx.x * 256 + threadIdx.x;
    const size_t total = (size_t)Bsz * LGl * NGd;
    if (idx >= total) return;
    int c = (int)(idx % NGd);
    int r = (int)((idx / NGd) % LGl);
    int b = (int)(idx / ((size_t)NGd * LGl));
    int n = r / 3, j = r % 3;
    float v;
    if (j == 0) v = zg[((size_t)(b * Nst + n)) * NGd + c];
    else        v = xg[((size_t)b * (2 * Nst) + 2 * n + (j - 1)) * NGd + c];
    gx[idx] = v;
}

// ---------------- extract xg_out rows (3n+1, 3n+2) ----------------------
__global__ void __launch_bounds__(256)
extract_kernel(const float* __restrict__ gx, float* __restrict__ out) {
    size_t idx = (size_t)blockIdx.x * 256 + threadIdx.x;
    const size_t total = (size_t)Bsz * 2 * Nst * NGd;
    if (idx >= total) return;
    int c = (int)(idx % NGd);
    int r = (int)((idx / NGd) % (2 * Nst));
    int b = (int)(idx / ((size_t)NGd * 2 * Nst));
    int n = r / 2, j = r % 2;
    out[idx] = gx[((size_t)b * LGl + 3 * n + 1 + j) * NGd + c];
}

// ---------------- host-side GEMM launcher -------------------------------
static void launch_gemm(const u16* A, const float* Bw, const float* bias,
                        const float* resid, float* outF, u16* outB,
                        int M, int N, int K, int gelu, hipStream_t s) {
    dim3 grid(N / GBN, M / GBM);
    gemm_bf16_kernel<<<grid, 256, 0, s>>>(A, Bw, bias, resid, outF, outB, M, N, K, gelu);
}

// ---------------- workspace arena (bytes) -------------------------------
static const size_t SZ_ACT32 = (size_t)T1 * NLd * sizeof(float);  // 50331648
static const size_t SZ_ACT16 = (size_t)T1 * NLd * sizeof(u16);    // 25165824
static const size_t OFF_X1 = 0;
static const size_t OFF_LN = OFF_X1 + SZ_ACT32;
static const size_t OFF_Q  = OFF_LN + SZ_ACT16;
static const size_t OFF_K  = OFF_Q  + SZ_ACT16;
static const size_t OFF_V  = OFF_K  + SZ_ACT16;
static const size_t OFF_Y  = OFF_V  + SZ_ACT16;
static const size_t OFF_H  = OFF_Q;                 // MLP hidden aliases q..y
static const size_t ARENA  = OFF_Y + SZ_ACT16;      // 176,160,768 bytes
// global-stage sub-arena inside dead X1 region:
static const size_t OFF_GX  = 0;                                  // f32 [3072,768]
static const size_t OFF_ZG  = OFF_GX + (size_t)T2 * NGd * 4;      // f32 [1024,768]
static const size_t OFF_GLN = OFF_ZG + (size_t)SEQ1 * NGd * 4;    // u16 [3072,768]
static const size_t OFF_GQ  = OFF_GLN + (size_t)T2 * NGd * 2;
static const size_t OFF_GK  = OFF_GQ  + (size_t)T2 * NGd * 2;
static const size_t OFF_GV  = OFF_GK  + (size_t)T2 * NGd * 2;
static const size_t OFF_GY  = OFF_GV  + (size_t)T2 * NGd * 2;

extern "C" void kernel_launch(void* const* d_in, const int* in_sizes, int n_in,
                              void* d_out, int out_size, void* d_ws, size_t ws_size,
                              hipStream_t stream) {
    (void)in_sizes; (void)n_in; (void)out_size;
    if (ws_size < ARENA) return;

    const float* xl      = (const float*)d_in[0];
    const float* xg      = (const float*)d_in[1];
    const float* l_ln1w  = (const float*)d_in[2];
    const float* l_ln1b  = (const float*)d_in[3];
    const float* l_ln2w  = (const float*)d_in[4];
    const float* l_ln2b  = (const float*)d_in[5];
    const float* l_wq    = (const float*)d_in[6];
    const float* l_bq    = (const float*)d_in[7];
    const float* l_wk    = (const float*)d_in[8];
    const float* l_bk    = (const float*)d_in[9];
    const float* l_wv    = (const float*)d_in[10];
    const float* l_bv    = (const float*)d_in[11];
    const float* l_wo    = (const float*)d_in[12];
    const float* l_bo    = (const float*)d_in[13];
    const float* l_w1    = (const float*)d_in[14];
    const float* l_b1    = (const float*)d_in[15];
    const float* l_w2    = (const float*)d_in[16];
    const float* l_b2    = (const float*)d_in[17];
    const float* g_ln1w  = (const float*)d_in[18];
    const float* g_ln1b  = (const float*)d_in[19];
    const float* g_ln2w  = (const float*)d_in[20];
    const float* g_ln2b  = (const float*)d_in[21];
    const float* g_wq    = (const float*)d_in[22];
    const float* g_bq    = (const float*)d_in[23];
    const float* g_wk    = (const float*)d_in[24];
    const float* g_bk    = (const float*)d_in[25];
    const float* g_wv    = (const float*)d_in[26];
    const float* g_bv    = (const float*)d_in[27];
    const float* g_wo    = (const float*)d_in[28];
    const float* g_bo    = (const float*)d_in[29];
    const float* g_w1    = (const float*)d_in[30];
    const float* g_b1    = (const float*)d_in[31];
    const float* g_w2    = (const float*)d_in[32];
    const float* g_b2    = (const float*)d_in[33];
    const float* d_wM    = (const float*)d_in[34];
    const float* d_bM    = (const float*)d_in[35];

    float* outXL = (float*)d_out;                                // [32,32,64,192]
    float* outXG = outXL + (size_t)T1 * NLd;                     // [32,64,768]

    char* ws = (char*)d_ws;
    float* X1 = (float*)(ws + OFF_X1);
    u16*   LN = (u16*)(ws + OFF_LN);
    u16*   Qb = (u16*)(ws + OFF_Q);
    u16*   Kb = (u16*)(ws + OFF_K);
    u16*   Vb = (u16*)(ws + OFF_V);
    u16*   Yb = (u16*)(ws + OFF_Y);
    u16*   Hb = (u16*)(ws + OFF_H);
    float* GX = (float*)(ws + OFF_GX);
    float* ZG = (float*)(ws + OFF_ZG);
    u16*   GLN = (u16*)(ws + OFF_GLN);
    u16*   GQ = (u16*)(ws + OFF_GQ);
    u16*   GK = (u16*)(ws + OFF_GK);
    u16*   GV = (u16*)(ws + OFF_GV);
    u16*   GY = (u16*)(ws + OFF_GY);
    u16*   GH = (u16*)(ws + OFF_H);

    // LDS bytes: (L*L + 3*L*hd) * 2
    const size_t ldsLocal  = ((size_t)Mloc * Mloc + 3 * (size_t)Mloc * 32) * 2;  // 20480
    const size_t ldsGlobal = ((size_t)LGl * LGl + 3 * (size_t)LGl * 64) * 2;     // 55296
    (void)hipFuncSetAttribute((const void*)attn_kernel,
                              hipFuncAttributeMaxDynamicSharedMemorySize,
                              (int)ldsGlobal);

    // ---------------- local block ----------------
    ln_bf16_kernel<<<T1, 256, 0, stream>>>(xl, l_ln1w, l_ln1b, LN, NLd);
    launch_gemm(LN, l_wq, l_bq, nullptr, nullptr, Qb, T1, NLd, NLd, 0, stream);
    launch_gemm(LN, l_wk, l_bk, nullptr, nullptr, Kb, T1, NLd, NLd, 0, stream);
    launch_gemm(LN, l_wv, l_bv, nullptr, nullptr, Vb, T1, NLd, NLd, 0, stream);
    attn_kernel<<<SEQ1 * HLh, 256, ldsLocal, stream>>>(Qb, Kb, Vb, Yb,
                                                       Mloc, 32, HLh,
                                                       0.17677669529663687f);
    launch_gemm(Yb, l_wo, l_bo, xl, X1, nullptr, T1, NLd, NLd, 0, stream);
    ln_bf16_kernel<<<T1, 256, 0, stream>>>(X1, l_ln2w, l_ln2b, LN, NLd);
    launch_gemm(LN, l_w1, l_b1, nullptr, nullptr, Hb, T1, 4 * NLd, NLd, 1, stream);
    launch_gemm(Hb, l_w2, l_b2, X1, outXL, LN, T1, NLd, 4 * NLd, 0, stream);

    // ---------------- group summary projection ----------------
    // LN now holds xl_out in bf16, viewed as [1024, 12288]
    launch_gemm(LN, d_wM, d_bM, nullptr, ZG, nullptr, SEQ1, NGd, Mloc * NLd, 0, stream);
    {
        size_t total = (size_t)Bsz * LGl * NGd;
        interleave_kernel<<<(unsigned)((total + 255) / 256), 256, 0, stream>>>(ZG, xg, GX);
    }

    // ---------------- global block ----------------
    ln_bf16_kernel<<<T2, 256, 0, stream>>>(GX, g_ln1w, g_ln1b, GLN, NGd);
    launch_gemm(GLN, g_wq, g_bq, nullptr, nullptr, GQ, T2, NGd, NGd, 0, stream);
    launch_gemm(GLN, g_wk, g_bk, nullptr, nullptr, GK, T2, NGd, NGd, 0, stream);
    launch_gemm(GLN, g_wv, g_bv, nullptr, nullptr, GV, T2, NGd, NGd, 0, stream);
    attn_kernel<<<Bsz * HGh, 256, ldsGlobal, stream>>>(GQ, GK, GV, GY,
                                                       LGl, 64, HGh, 0.125f);
    launch_gemm(GY, g_wo, g_bo, GX, GX, nullptr, T2, NGd, NGd, 0, stream);
    ln_bf16_kernel<<<T2, 256, 0, stream>>>(GX, g_ln2w, g_ln2b, GLN, NGd);
    launch_gemm(GLN, g_w1, g_b1, nullptr, nullptr, GH, T2, 4 * NGd, NGd, 1, stream);
    launch_gemm(GH, g_w2, g_b2, GX, GX, nullptr, T2, NGd, 4 * NGd, 0, stream);
    {
        size_t total = (size_t)Bsz * 2 * Nst * NGd;
        extract_kernel<<<(unsigned)((total + 255) / 256), 256, 0, stream>>>(GX, outXG);
    }
}
